// Correlation_15719580304324
// MI455X (gfx1250) — matrix operand order repaired and checked by reading, not compiled, not run on hardware
//
#include <hip/hip_runtime.h>

// CDNA5 / gfx1250 vector types for WMMA
typedef __attribute__((ext_vector_type(16))) __bf16 v16bf;
typedef __attribute__((ext_vector_type(8)))  __bf16 v8bf;
typedef __attribute__((ext_vector_type(2)))  __bf16 v2bf;
typedef __attribute__((ext_vector_type(8)))  float  v8f;

#define C_  256
#define H_  96
#define W_  160
#define NQ_ 81
#define AP  264   // LDS pitch (bf16 elems): 256 + 8 pad -> 132-dword row stride, bank-friendly
#define BP  264

// Block: 96 threads = 3 waves. Wave w covers dy in {3w-4, 3w-3, 3w-2}.
// Each block handles one (n, y, 16-wide x tile); per (wave,dy) we compute the
// Gram matrix between the A tile (16 x-cols x 256 ch) and a 32-col B slab of
// data2 row y+dy via two v_wmma_f32_16x16x32_bf16 chains, and keep the
// diagonal band dx in [-4,4].
__global__ __launch_bounds__(96) void corr_wmma_kernel(
    const float* __restrict__ d1, const float* __restrict__ d2,
    float* __restrict__ out) {
  __shared__ __bf16 Ash[16 * AP];
  __shared__ __bf16 Bsh[3][32 * BP];

  const int x0   = blockIdx.x * 16;
  const int y    = blockIdx.y;
  const int n    = blockIdx.z;
  const int tid  = threadIdx.x;
  const int wave = tid >> 5;
  const int lane = tid & 31;
  const int half = lane >> 4;   // lane group 0/1
  const int nidx = lane & 15;   // row (M) for A, column (N) for B/D

  const size_t chstride = (size_t)H_ * W_;  // channel stride in elements

  // ---- stage A: data1[n, :, y, x0..x0+15] -> Ash[x][c] as bf16 ----
  {
    const int xl  = tid & 15;
    const int cp0 = tid >> 4;   // 0..5
    const float* g = d1 + ((size_t)n * C_) * chstride + (size_t)y * W_ + (x0 + xl);
    for (int cp = cp0; cp < C_ / 2; cp += 6) {
      const int c = cp * 2;
      float f0 = g[(size_t)c * chstride];
      float f1 = g[(size_t)(c + 1) * chstride];
      v2bf p; p[0] = (__bf16)f0; p[1] = (__bf16)f1;
      *(v2bf*)&Ash[xl * AP + c] = p;
    }
  }
  __syncthreads();

  __bf16* Bw = Bsh[wave];

  for (int s = 0; s < 3; ++s) {
    const int dy = wave * 3 + s - 4;
    const int yy = y + dy;

    // ---- stage B: data2[n, :, yy, x0-4 .. x0+27] -> Bw[j][c] ----
    // Branch-free OOB handling: loads are unconditional from a clamped
    // (always-valid) address; one 0.0/1.0 mask multiply zeroes OOB lanes.
    // This keeps the loop a clean global_load -> cvt_pk -> ds_store stream
    // with no per-element EXEC manipulation.
    {
      const int j  = lane;            // column 0..31 <-> x' = x0-4+j
      const int xg = x0 - 4 + j;
      const bool inb = (yy >= 0) && (yy < H_) && (xg >= 0) && (xg < W_);
      const float msk = inb ? 1.0f : 0.0f;
      int yc = yy < 0 ? 0 : (yy >= H_ ? H_ - 1 : yy);
      int xc = xg < 0 ? 0 : (xg >= W_ ? W_ - 1 : xg);
      const float* g = d2 + ((size_t)n * C_) * chstride + (size_t)yc * W_ + xc;
      #pragma unroll 8
      for (int c = 0; c < C_; c += 2) {
        float f0 = g[(size_t)c * chstride] * msk;
        float f1 = g[(size_t)(c + 1) * chstride] * msk;
        v2bf p; p[0] = (__bf16)f0; p[1] = (__bf16)f1;
        *(v2bf*)&Bw[j * BP + c] = p;
      }
    }
    // Wave-synchronous LDS: DS ops are in-order within a wave; compiler
    // inserts the s_wait_dscnt for the store->load dependence below.

    v8f acc0 = {0.f, 0.f, 0.f, 0.f, 0.f, 0.f, 0.f, 0.f};
    v8f acc1 = {0.f, 0.f, 0.f, 0.f, 0.f, 0.f, 0.f, 0.f};

    #pragma unroll
    for (int k0 = 0; k0 < C_; k0 += 32) {
      // A fragment: lane group 0 holds K 0..15, group 1 K 16..31
      v8bf alo = *(const v8bf*)&Ash[nidx * AP + k0 + half * 16];
      v8bf ahi = *(const v8bf*)&Ash[nidx * AP + k0 + half * 16 + 8];
      v16bf a;
      #pragma unroll
      for (int i = 0; i < 8; ++i) { a[i] = alo[i]; a[8 + i] = ahi[i]; }
      // B fragments: lanes 0-15 hold K 0..15, lanes 16-31 hold K 16..31
      v16bf b0 = *(const v16bf*)&Bw[nidx * BP + k0 + half * 16];
      v16bf b1 = *(const v16bf*)&Bw[(16 + nidx) * BP + k0 + half * 16];
      acc0 = __builtin_amdgcn_wmma_f32_16x16x32_bf16(
          false, a, false, b0, (short)0, acc0, false, false);
      acc1 = __builtin_amdgcn_wmma_f32_16x16x32_bf16(
          false, a, false, b1, (short)0, acc1, false, false);
    }

    // ---- band extraction: D vgpr v <-> M = v + 8*half, lane <-> N ----
    #pragma unroll
    for (int v = 0; v < 8; ++v) {
      const int m = v + half * 8;     // output x = x0 + m
      {
        const int dx = nidx - m - 4;          // tile0: x' = x0 - 4 + nidx
        if (dx >= -4 && dx <= 4) {
          const int q = (dy + 4) * 9 + (dx + 4);
          out[(((size_t)n * NQ_ + q) * H_ + y) * W_ + (x0 + m)] =
              acc0[v] * (1.0f / 256.0f);
        }
      }
      {
        const int dx = nidx + 12 - m;         // tile1: x' = x0 + 12 + nidx
        if (dx >= -4 && dx <= 4) {
          const int q = (dy + 4) * 9 + (dx + 4);
          out[(((size_t)n * NQ_ + q) * H_ + y) * W_ + (x0 + m)] =
              acc1[v] * (1.0f / 256.0f);
        }
      }
    }
  }
}

extern "C" void kernel_launch(void* const* d_in, const int* in_sizes, int n_in,
                              void* d_out, int out_size, void* d_ws, size_t ws_size,
                              hipStream_t stream) {
  const float* d1 = (const float*)d_in[0];
  const float* d2 = (const float*)d_in[1];
  float* out = (float*)d_out;
  (void)in_sizes; (void)n_in; (void)out_size; (void)d_ws; (void)ws_size;
  dim3 grid(W_ / 16, H_, 8);   // 10 x-tiles, 96 rows, 8 batch
  corr_wmma_kernel<<<grid, 96, 0, stream>>>(d1, d2, out);
}
